// MultiHeadAttn_91225105367259
// MI455X (gfx1250) — matrix operand order repaired and checked by reading, not compiled, not run on hardware
//
#include <hip/hip_runtime.h>

// ---------------------------------------------------------------------------
// MHA for MI455X (gfx1250): bf16 WMMA GEMMs + flash-attention.
// B=4, N=2048, D=1024, H=16, DH=64, OUT=1024.
// fp32 -> bf16 conversion hoisted to a one-shot pass; GEMMs move only bf16.
// Wave tile 32x64 (2 A-frags x 4 B-frags -> 8 WMMAs per k-step) with all
// fragments in distinct registers so LDS loads pipeline against WMMAs.
// ---------------------------------------------------------------------------

typedef __attribute__((ext_vector_type(16))) __bf16 v16bf;
typedef __attribute__((ext_vector_type(8)))  __bf16 v8bf;    // 16 bytes
typedef __attribute__((ext_vector_type(4)))  __bf16 v4bf;    // 8 bytes
typedef __attribute__((ext_vector_type(8)))  float  v8f;
typedef __attribute__((ext_vector_type(4)))  float  f4;      // 16 bytes

#define DEV __device__ __forceinline__

constexpr int Bc   = 4;
constexpr int Nc   = 2048;
constexpr int Dc   = 1024;
constexpr int Hc   = 16;
constexpr int DHc  = 64;
constexpr int OUTc = 1024;
constexpr int Mrows = Bc * Nc;   // 8192

DEV v8f wmma_bf16(v16bf a, v16bf b, v8f c) {
    return __builtin_amdgcn_wmma_f32_16x16x32_bf16(
        false, a, false, b, (short)0, c, false, false);
}

DEV v16bf join16(v8bf lo, v8bf hi) {
    return __builtin_shufflevector(lo, hi, 0, 1, 2, 3, 4, 5, 6, 7,
                                           8, 9, 10, 11, 12, 13, 14, 15);
}

// A-fragment (16x32 bf16): lane half hf reads two 16-byte runs at
// k = 16*hf and k = 16*hf + 8 (row stride must be a multiple of 8 bf16).
DEV v16bf loadA(const __bf16* row, int hf) {
    v8bf lo = *(const v8bf*)(row + 16 * hf);
    v8bf hi = *(const v8bf*)(row + 16 * hf + 8);
    return join16(lo, hi);
}

// Contiguous 16-element fragment (fragment-order staging or K rows).
DEV v16bf load16(const __bf16* p) {
    v8bf lo = *(const v8bf*)(p);
    v8bf hi = *(const v8bf*)(p + 8);
    return join16(lo, hi);
}

// ---------------------------------------------------------------------------
// One-shot fp32 -> bf16 conversion (vectorized, native cvt).
// ---------------------------------------------------------------------------
__global__ __launch_bounds__(256)
void cvt_f32_bf16_kernel(const float* __restrict__ src,
                         __bf16* __restrict__ dst, int n4)
{
    int i = blockIdx.x * 256 + threadIdx.x;
    if (i < n4) {
        f4 x = *(const f4*)(src + 4 * (size_t)i);
        *(v4bf*)(dst + 4 * (size_t)i) = __builtin_convertvector(x, v4bf);
    }
}

// ---------------------------------------------------------------------------
// Kernel 1: X[8192,1024] bf16 @ W[1024,1024] bf16 + bias, scaled, bf16 out
// reordered to [B,H,N,DH].  Block tile 128x128, K chunk 64, wave tile 32x64.
// ---------------------------------------------------------------------------
__global__ __launch_bounds__(256)
void proj_qkv_kernel(const __bf16* __restrict__ X, const __bf16* __restrict__ W,
                     const float* __restrict__ bias, float scale,
                     __bf16* __restrict__ out)
{
    __shared__ __bf16 Xs[128][72];            // 128 rows x 64 k (stride 144B)
    __shared__ __bf16 Wfrag[2][8][32][16];    // [kStep][nTile][lane][j]

    const int tid  = threadIdx.x;
    const int lane = tid & 31;
    const int wv   = tid >> 5;
    const int ln   = lane & 15;
    const int hf   = lane >> 4;
    const int waveM = wv & 3;                 // rows 32*waveM .. +31
    const int waveN = wv >> 2;                // cols 64*waveN .. +63
    const int rowBase = blockIdx.y * 128;
    const int nBase   = blockIdx.x * 128;

    v8f acc[2][4] = {};

    for (int k0 = 0; k0 < Dc; k0 += 64) {
        __syncthreads();
        // ---- X chunk: 128x64 bf16, 16B vector loads/stores --------------
#pragma unroll
        for (int i = 0; i < 4; ++i) {
            int idx = tid + 256 * i;             // 0..1023 vectors
            int r = idx >> 3, q = idx & 7;
            *(v8bf*)&Xs[r][8 * q] =
                *(const v8bf*)(X + (size_t)(rowBase + r) * Dc + k0 + 8 * q);
        }
        // ---- W chunk: 64x128 bf16 -> fragment order ---------------------
#pragma unroll
        for (int i = 0; i < 4; ++i) {
            int idx = tid + 256 * i;             // 0..1023 vectors
            int r = idx >> 4, q = idx & 15;      // k row 0..63, n/8
            v8bf w = *(const v8bf*)(W + (size_t)(k0 + r) * Dc + nBase + 8 * q);
            int st = r >> 5, rr = r & 31;
            int whf = rr >> 4, wj = rr & 15;
            int c   = q >> 1;
            int ln0 = 8 * (q & 1);
#pragma unroll
            for (int e = 0; e < 8; ++e)
                Wfrag[st][c][ln0 + e + 16 * whf][wj] = w[e];
        }
        __syncthreads();

#pragma unroll
        for (int st = 0; st < 2; ++st) {
            v16bf a0 = loadA(&Xs[32 * waveM + ln][32 * st], hf);
            v16bf a1 = loadA(&Xs[32 * waveM + 16 + ln][32 * st], hf);
            v16bf b0 = load16(&Wfrag[st][4 * waveN + 0][lane][0]);
            v16bf b1 = load16(&Wfrag[st][4 * waveN + 1][lane][0]);
            v16bf b2 = load16(&Wfrag[st][4 * waveN + 2][lane][0]);
            v16bf b3 = load16(&Wfrag[st][4 * waveN + 3][lane][0]);
            acc[0][0] = wmma_bf16(a0, b0, acc[0][0]);
            acc[0][1] = wmma_bf16(a0, b1, acc[0][1]);
            acc[0][2] = wmma_bf16(a0, b2, acc[0][2]);
            acc[0][3] = wmma_bf16(a0, b3, acc[0][3]);
            acc[1][0] = wmma_bf16(a1, b0, acc[1][0]);
            acc[1][1] = wmma_bf16(a1, b1, acc[1][1]);
            acc[1][2] = wmma_bf16(a1, b2, acc[1][2]);
            acc[1][3] = wmma_bf16(a1, b3, acc[1][3]);
        }
    }

    float bs[4];
#pragma unroll
    for (int c = 0; c < 4; ++c)
        bs[c] = bias[nBase + 64 * waveN + 16 * c + ln];

#pragma unroll
    for (int mi = 0; mi < 2; ++mi) {
#pragma unroll
        for (int c = 0; c < 4; ++c) {
#pragma unroll
            for (int r = 0; r < 8; ++r) {
                int m    = 32 * waveM + 16 * mi + r + 8 * hf;
                int grow = rowBase + m;
                int ncol = nBase + 64 * waveN + 16 * c + ln;
                float val = (acc[mi][c][r] + bs[c]) * scale;
                int bidx = grow >> 11;
                int nrow = grow & (Nc - 1);
                int h  = ncol >> 6;
                int dh = ncol & (DHc - 1);
                out[((size_t)(bidx * Hc + h) * Nc + nrow) * DHc + dh] =
                    static_cast<__bf16>(val);
            }
        }
    }
}

// ---------------------------------------------------------------------------
// Kernel 2: flash attention.  One block per (head, 128 Q rows).
// ---------------------------------------------------------------------------
__global__ __launch_bounds__(256)
void attn_kernel(const __bf16* __restrict__ qb, const __bf16* __restrict__ kb,
                 const __bf16* __restrict__ vb, __bf16* __restrict__ ctx)
{
    __shared__ __bf16 Qs[128][72];          // stride 144B = 9*16 (aligned)
    __shared__ __bf16 Ks[64][72];           // row-contiguous (B-frag of K^T)
    __shared__ __bf16 Vfrag[2][4][32][16];  // [kStep][dhTile][lane][j]
    __shared__ __bf16 Ps[8][16][72];        // per-wave P staging

    const int tid  = threadIdx.x;
    const int lane = tid & 31;
    const int wv   = tid >> 5;
    const int ln   = lane & 15;
    const int hf   = lane >> 4;
    const int bh    = blockIdx.y;
    const int qBase = blockIdx.x * 128;

    const __bf16* qp = qb + (size_t)bh * Nc * DHc;
    const __bf16* kp = kb + (size_t)bh * Nc * DHc;
    const __bf16* vp = vb + (size_t)bh * Nc * DHc;

#pragma unroll
    for (int i = 0; i < 4; ++i) {           // Q tile: 1024 x 16B vectors
        int idx = tid + 256 * i;
        int r = idx >> 3, q = idx & 7;
        *(v8bf*)&Qs[r][8 * q] =
            *(const v8bf*)(qp + (size_t)(qBase + r) * DHc + 8 * q);
    }

    v8f o[4] = {};
    float m_run[8], l_run[8];
#pragma unroll
    for (int r = 0; r < 8; ++r) { m_run[r] = -1e30f; l_run[r] = 0.f; }

    for (int kt = 0; kt < Nc / 64; ++kt) {
        __syncthreads();
#pragma unroll
        for (int i = 0; i < 2; ++i) {       // K tile: 512 x 16B vectors
            int idx = tid + 256 * i;
            int r = idx >> 3, q = idx & 7;
            *(v8bf*)&Ks[r][8 * q] =
                *(const v8bf*)(kp + (size_t)(kt * 64 + r) * DHc + 8 * q);
        }
#pragma unroll
        for (int i = 0; i < 2; ++i) {       // V tile -> fragment order
            int idx = tid + 256 * i;
            int key = idx >> 3, q = idx & 7;
            v8bf v = *(const v8bf*)(vp + (size_t)(kt * 64 + key) * DHc + 8 * q);
            int st = key >> 5, vhf = (key >> 4) & 1, vj = key & 15;
            int c = q >> 1;
            int ln0 = 8 * (q & 1);
#pragma unroll
            for (int e = 0; e < 8; ++e)
                Vfrag[st][c][ln0 + e + 16 * vhf][vj] = v[e];
        }
        __syncthreads();

        if (kt + 1 < Nc / 64) {
            __builtin_prefetch(kp + (size_t)((kt + 1) * 64 + (tid >> 2)) * DHc, 0, 1);
            __builtin_prefetch(vp + (size_t)((kt + 1) * 64 + (tid >> 2)) * DHc, 0, 1);
        }

        // ---- S = Q * K^T : 16 rows x 64 keys per wave -------------------
        v8f s[4] = {};
        {
            v16bf qa0 = loadA(&Qs[16 * wv + ln][0], hf);
            v16bf qa1 = loadA(&Qs[16 * wv + ln][32], hf);
            v16bf b0 = load16(&Ks[ln][16 * hf]);
            v16bf b1 = load16(&Ks[16 + ln][16 * hf]);
            v16bf b2 = load16(&Ks[32 + ln][16 * hf]);
            v16bf b3 = load16(&Ks[48 + ln][16 * hf]);
            s[0] = wmma_bf16(qa0, b0, s[0]);
            s[1] = wmma_bf16(qa0, b1, s[1]);
            s[2] = wmma_bf16(qa0, b2, s[2]);
            s[3] = wmma_bf16(qa0, b3, s[3]);
            b0 = load16(&Ks[ln][32 + 16 * hf]);
            b1 = load16(&Ks[16 + ln][32 + 16 * hf]);
            b2 = load16(&Ks[32 + ln][32 + 16 * hf]);
            b3 = load16(&Ks[48 + ln][32 + 16 * hf]);
            s[0] = wmma_bf16(qa1, b0, s[0]);
            s[1] = wmma_bf16(qa1, b1, s[1]);
            s[2] = wmma_bf16(qa1, b2, s[2]);
            s[3] = wmma_bf16(qa1, b3, s[3]);
        }

        // ---- online softmax ---------------------------------------------
        float corr[8];
#pragma unroll
        for (int r = 0; r < 8; ++r) {
            float mx = fmaxf(fmaxf(s[0][r], s[1][r]), fmaxf(s[2][r], s[3][r]));
            mx = fmaxf(mx, __shfl_xor(mx, 1, 32));
            mx = fmaxf(mx, __shfl_xor(mx, 2, 32));
            mx = fmaxf(mx, __shfl_xor(mx, 4, 32));
            mx = fmaxf(mx, __shfl_xor(mx, 8, 32));
            float nm = fmaxf(m_run[r], mx);
            corr[r]  = __expf(m_run[r] - nm);
            m_run[r] = nm;
            float rs = 0.f;
#pragma unroll
            for (int c = 0; c < 4; ++c) {
                float p = __expf(s[c][r] - nm);
                s[c][r] = p;
                rs += p;
            }
            rs += __shfl_xor(rs, 1, 32);
            rs += __shfl_xor(rs, 2, 32);
            rs += __shfl_xor(rs, 4, 32);
            rs += __shfl_xor(rs, 8, 32);
            l_run[r] = l_run[r] * corr[r] + rs;
#pragma unroll
            for (int c = 0; c < 4; ++c) {
                o[c][r] *= corr[r];
                Ps[wv][r + 8 * hf][16 * c + ln] = static_cast<__bf16>(s[c][r]);
            }
        }
        // same-wave DS ordering keeps Ps stores visible to loads below

        // ---- O += P * V -------------------------------------------------
        {
            v16bf pa0 = loadA(&Ps[wv][ln][0], hf);
            v16bf pa1 = loadA(&Ps[wv][ln][32], hf);
            v16bf b0 = load16(&Vfrag[0][0][lane][0]);
            v16bf b1 = load16(&Vfrag[0][1][lane][0]);
            v16bf b2 = load16(&Vfrag[0][2][lane][0]);
            v16bf b3 = load16(&Vfrag[0][3][lane][0]);
            o[0] = wmma_bf16(pa0, b0, o[0]);
            o[1] = wmma_bf16(pa0, b1, o[1]);
            o[2] = wmma_bf16(pa0, b2, o[2]);
            o[3] = wmma_bf16(pa0, b3, o[3]);
            b0 = load16(&Vfrag[1][0][lane][0]);
            b1 = load16(&Vfrag[1][1][lane][0]);
            b2 = load16(&Vfrag[1][2][lane][0]);
            b3 = load16(&Vfrag[1][3][lane][0]);
            o[0] = wmma_bf16(pa1, b0, o[0]);
            o[1] = wmma_bf16(pa1, b1, o[1]);
            o[2] = wmma_bf16(pa1, b2, o[2]);
            o[3] = wmma_bf16(pa1, b3, o[3]);
        }
    }

    const int bidx = bh >> 4, h = bh & 15;
#pragma unroll
    for (int c = 0; c < 4; ++c) {
#pragma unroll
        for (int r = 0; r < 8; ++r) {
            int m    = 16 * wv + r + 8 * hf;
            int nrow = qBase + m;
            float val = o[c][r] / l_run[r];
            ctx[(size_t)(bidx * Nc + nrow) * Dc + h * DHc + 16 * c + ln] =
                static_cast<__bf16>(val);
        }
    }
}

// ---------------------------------------------------------------------------
// Kernel 3: out = ctx(bf16)[8192,1024] @ Wo(bf16)[1024,1024] + bo, fp32 out.
// Block tile 128x128, K chunk 64, wave tile 32x64.
// ---------------------------------------------------------------------------
__global__ __launch_bounds__(256)
void out_proj_kernel(const __bf16* __restrict__ A, const __bf16* __restrict__ W,
                     const float* __restrict__ bias, float* __restrict__ out)
{
    __shared__ __bf16 Xs[128][72];
    __shared__ __bf16 Wfrag[2][8][32][16];

    const int tid  = threadIdx.x;
    const int lane = tid & 31;
    const int wv   = tid >> 5;
    const int ln   = lane & 15;
    const int hf   = lane >> 4;
    const int waveM = wv & 3;
    const int waveN = wv >> 2;
    const int rowBase = blockIdx.y * 128;
    const int nBase   = blockIdx.x * 128;

    v8f acc[2][4] = {};

    for (int k0 = 0; k0 < Dc; k0 += 64) {
        __syncthreads();
#pragma unroll
        for (int i = 0; i < 4; ++i) {
            int idx = tid + 256 * i;
            int r = idx >> 3, q = idx & 7;
            *(v8bf*)&Xs[r][8 * q] =
                *(const v8bf*)(A + (size_t)(rowBase + r) * Dc + k0 + 8 * q);
        }
#pragma unroll
        for (int i = 0; i < 4; ++i) {
            int idx = tid + 256 * i;
            int r = idx >> 4, q = idx & 15;
            v8bf w = *(const v8bf*)(W + (size_t)(k0 + r) * OUTc + nBase + 8 * q);
            int st = r >> 5, rr = r & 31;
            int whf = rr >> 4, wj = rr & 15;
            int c   = q >> 1;
            int ln0 = 8 * (q & 1);
#pragma unroll
            for (int e = 0; e < 8; ++e)
                Wfrag[st][c][ln0 + e + 16 * whf][wj] = w[e];
        }
        __syncthreads();

#pragma unroll
        for (int st = 0; st < 2; ++st) {
            v16bf a0 = loadA(&Xs[32 * waveM + ln][32 * st], hf);
            v16bf a1 = loadA(&Xs[32 * waveM + 16 + ln][32 * st], hf);
            v16bf b0 = load16(&Wfrag[st][4 * waveN + 0][lane][0]);
            v16bf b1 = load16(&Wfrag[st][4 * waveN + 1][lane][0]);
            v16bf b2 = load16(&Wfrag[st][4 * waveN + 2][lane][0]);
            v16bf b3 = load16(&Wfrag[st][4 * waveN + 3][lane][0]);
            acc[0][0] = wmma_bf16(a0, b0, acc[0][0]);
            acc[0][1] = wmma_bf16(a0, b1, acc[0][1]);
            acc[0][2] = wmma_bf16(a0, b2, acc[0][2]);
            acc[0][3] = wmma_bf16(a0, b3, acc[0][3]);
            acc[1][0] = wmma_bf16(a1, b0, acc[1][0]);
            acc[1][1] = wmma_bf16(a1, b1, acc[1][1]);
            acc[1][2] = wmma_bf16(a1, b2, acc[1][2]);
            acc[1][3] = wmma_bf16(a1, b3, acc[1][3]);
        }
    }

    float bs[4];
#pragma unroll
    for (int c = 0; c < 4; ++c)
        bs[c] = bias[nBase + 64 * waveN + 16 * c + ln];

#pragma unroll
    for (int mi = 0; mi < 2; ++mi) {
#pragma unroll
        for (int c = 0; c < 4; ++c) {
#pragma unroll
            for (int r = 0; r < 8; ++r) {
                int m    = 32 * waveM + 16 * mi + r + 8 * hf;
                int grow = rowBase + m;
                int ncol = nBase + 64 * waveN + 16 * c + ln;
                out[(size_t)grow * OUTc + ncol] = acc[mi][c][r] + bs[c];
            }
        }
    }
}

// ---------------------------------------------------------------------------
extern "C" void kernel_launch(void* const* d_in, const int* in_sizes, int n_in,
                              void* d_out, int out_size, void* d_ws, size_t ws_size,
                              hipStream_t stream)
{
    const float* keyX = (const float*)d_in[0];
    const float* valX = (const float*)d_in[1];
    const float* qryX = (const float*)d_in[2];
    const float* Wk   = (const float*)d_in[3];
    const float* bk   = (const float*)d_in[4];
    const float* Wv   = (const float*)d_in[5];
    const float* bv   = (const float*)d_in[6];
    const float* Wq   = (const float*)d_in[7];
    const float* bq   = (const float*)d_in[8];
    const float* Wo   = (const float*)d_in[9];
    const float* bo   = (const float*)d_in[10];
    float* out = (float*)d_out;

    char* ws = (char*)d_ws;
    const size_t seg = (size_t)Mrows * Dc * sizeof(__bf16);   // 16 MB
    __bf16* qbuf   = (__bf16*)(ws);
    __bf16* kbuf   = (__bf16*)(ws + seg);
    __bf16* vbuf   = (__bf16*)(ws + 2 * seg);
    __bf16* cbuf   = (__bf16*)(ws + 3 * seg);
    __bf16* xstage = (__bf16*)(ws + 4 * seg);                 // reused 3x
    __bf16* wstage = (__bf16*)(ws + 5 * seg);                 // reused 4x

    const int actN4 = Mrows * Dc / 4;   // 2M vectors
    const int wN4   = Dc * Dc / 4;      // 256K vectors

    dim3 blk(256);
    dim3 gCvtA((actN4 + 255) / 256);
    dim3 gCvtW((wN4 + 255) / 256);
    dim3 gGemm(Dc / 128, Mrows / 128);     // (8, 64)

    // Q = (query @ Wq + bq) / 8, reordered [B,H,N,DH]
    cvt_f32_bf16_kernel<<<gCvtA, blk, 0, stream>>>(qryX, xstage, actN4);
    cvt_f32_bf16_kernel<<<gCvtW, blk, 0, stream>>>(Wq, wstage, wN4);
    proj_qkv_kernel<<<gGemm, blk, 0, stream>>>(xstage, wstage, bq, 0.125f, qbuf);

    cvt_f32_bf16_kernel<<<gCvtA, blk, 0, stream>>>(keyX, xstage, actN4);
    cvt_f32_bf16_kernel<<<gCvtW, blk, 0, stream>>>(Wk, wstage, wN4);
    proj_qkv_kernel<<<gGemm, blk, 0, stream>>>(xstage, wstage, bk, 1.0f, kbuf);

    cvt_f32_bf16_kernel<<<gCvtA, blk, 0, stream>>>(valX, xstage, actN4);
    cvt_f32_bf16_kernel<<<gCvtW, blk, 0, stream>>>(Wv, wstage, wN4);
    proj_qkv_kernel<<<gGemm, blk, 0, stream>>>(xstage, wstage, bv, 1.0f, vbuf);

    dim3 gAttn(Nc / 128, Bc * Hc);         // (16, 64)
    attn_kernel<<<gAttn, blk, 0, stream>>>(qbuf, kbuf, vbuf, cbuf);

    cvt_f32_bf16_kernel<<<gCvtW, blk, 0, stream>>>(Wo, wstage, wN4);
    dim3 gOut(OUTc / 128, Mrows / 128);    // (8, 64)
    out_proj_kernel<<<gOut, blk, 0, stream>>>(cbuf, wstage, bo, out);
}
